// NSF_AR_22711787061432
// MI455X (gfx1250) — matrix operand order repaired and checked by reading, not compiled, not run on hardware
//
#include <hip/hip_runtime.h>
#include <hip/hip_bf16.h>
#include <math.h>

typedef __bf16        v16bf __attribute__((ext_vector_type(16)));
typedef float         v8f   __attribute__((ext_vector_type(8)));
typedef unsigned int  v4u   __attribute__((ext_vector_type(4)));

union Frag { v16bf v; v4u u[2]; };

#define BATCH  8192
#define DIMS   16
#define KBINS  32
#define HID    800
#define NT2    50        // HID/16 n-tiles
#define KT2    25        // HID/32 k-tiles
#define NOUT   95        // 3*K-1
#define NOUTP  96        // padded to 6 n-tiles
#define NT3    6
#define MROWS  64        // batch rows per block (4 WMMA M-tiles)
#define BTAIL  3.0f
#define PI_F   3.14159265358979323846f

// ---- d_ws layout (bf16 elements for packed weights, then f32 ld scratch) ----
#define SZ_B1  (16ull * NT2 * 512ull)                 // 409,600 bf16
#define SZ_B2  (16ull * KT2 * NT2 * 512ull)           // 10,240,000 bf16
#define SZ_B3  (16ull * KT2 * NT3 * 512ull)           // 1,228,800 bf16
#define OFF_B1 0ull
#define OFF_B2 (OFF_B1 + SZ_B1)
#define OFF_B3 (OFF_B2 + SZ_B2)
#define OFF_LD_BYTES ((OFF_B3 + SZ_B3) * 2ull)
#define WS_BYTES (OFF_LD_BYTES + (size_t)DIMS * BATCH * 4ull)

// ---- dynamic LDS layout (bytes) ----
#define LOFF_F    0u                                   // 64*32*2   = 4096
#define LOFF_H1   4096u                                // 64*800*2  = 102400
#define LOFF_H2   106496u                              // 64*800*2  = 102400
#define LOFF_OUT  208896u                              // 64*96*4   = 24576
#define LDS_BYTES 233472u                              // fits 320KB/WGP

__device__ __forceinline__ unsigned short f2bf(float f) {
    unsigned int u = __builtin_bit_cast(unsigned int, f);
    unsigned int r = u + 0x7FFFu + ((u >> 16) & 1u);   // round-to-nearest-even
    return (unsigned short)(r >> 16);
}

__device__ __forceinline__ float softplusf(float v) {
    return (v > 20.f) ? v : log1pf(expf(v));
}

// ---------------- weight repack kernels (f32 -> bf16, WMMA B-fragment order) -
// B fragment (32x16, bf16): lane = (kh<<4)|n ; element e -> K = kh*16+e, col n.
__global__ void pack_w1(const float* __restrict__ W1, unsigned short* __restrict__ B1) {
    unsigned idx = blockIdx.x * 256u + threadIdx.x;
    if (idx >= (unsigned)SZ_B1) return;
    unsigned e = idx & 15u, lane = (idx >> 4) & 31u, rest = idx >> 9;
    unsigned nt = rest % NT2, i = rest / NT2;
    unsigned n = lane & 15u, kh = lane >> 4;
    unsigned K = kh * 16u + e;                                   // K in [0,32)
    B1[idx] = f2bf(W1[((size_t)i * 32u + K) * HID + nt * 16u + n]);
}

__global__ void pack_w2(const float* __restrict__ W2, unsigned short* __restrict__ B2) {
    unsigned idx = blockIdx.x * 256u + threadIdx.x;
    if (idx >= (unsigned)SZ_B2) return;
    unsigned e = idx & 15u, lane = (idx >> 4) & 31u, rest = idx >> 9;
    unsigned nt = rest % NT2, rest2 = rest / NT2;
    unsigned kb = rest2 % KT2, i = rest2 / KT2;
    unsigned n = lane & 15u, kh = lane >> 4;
    unsigned K = kb * 32u + kh * 16u + e;
    B2[idx] = f2bf(W2[((size_t)i * HID + K) * HID + nt * 16u + n]);
}

__global__ void pack_w3(const float* __restrict__ W3, unsigned short* __restrict__ B3) {
    unsigned idx = blockIdx.x * 256u + threadIdx.x;
    if (idx >= (unsigned)SZ_B3) return;
    unsigned e = idx & 15u, lane = (idx >> 4) & 31u, rest = idx >> 9;
    unsigned nt = rest % NT3, rest2 = rest / NT3;
    unsigned kb = rest2 % KT2, i = rest2 / KT2;
    unsigned n = lane & 15u, kh = lane >> 4;
    unsigned K = kb * 32u + kh * 16u + e;
    unsigned col = nt * 16u + n;
    float v = (col < NOUT) ? W3[((size_t)i * HID + K) * NOUT + col] : 0.f;
    B3[idx] = f2bf(v);
}

// ---------------- fused MLP + spline kernel --------------------------------
// grid = (BATCH/64, DIMS), block = 256 threads = 8 waves (wave32)
// 64 batch rows per block: every 1KB B-fragment feeds 4 WMMAs (4x the FLOP/byte
// of an M=16 tiling), and A fragments are shared across an N-tile pair.
__global__ __launch_bounds__(256) void nsf_main(
    const float* __restrict__ x,
    const float* __restrict__ b1, const float* __restrict__ b2, const float* __restrict__ b3,
    const unsigned short* __restrict__ B1, const unsigned short* __restrict__ B2,
    const unsigned short* __restrict__ B3,
    float* __restrict__ z_out, float* __restrict__ ld_part)
{
    const int r0   = blockIdx.x * MROWS;
    const int i    = blockIdx.y;
    const int tid  = threadIdx.x;
    const int lane = tid & 31, wave = tid >> 5;
    const int n    = lane & 15, kh = lane >> 4;

    extern __shared__ __align__(16) char smem[];
    unsigned short* sF  = (unsigned short*)(smem + LOFF_F);   // [64][32]
    unsigned short* sH1 = (unsigned short*)(smem + LOFF_H1);  // [64][800]
    unsigned short* sH2 = (unsigned short*)(smem + LOFF_H2);  // [64][800]
    float*          sOUT = (float*)(smem + LOFF_OUT);         // [64][96]
    float*          sSp  = (float*)(smem + LOFF_H1);          // spline scratch (H1 dead)

    // ---- trig features: F[m][k] (K padded to 32 with zeros) ----
    const int ii = (i == 0) ? 1 : i;
    for (int idx = tid; idx < MROWS * 32; idx += 256) {
        int m = idx >> 5, k = idx & 31;
        float f = 0.f;
        if (k < 2 * ii) {
            int   c = (k < ii) ? k : (k - ii);
            float v = (i == 0) ? 0.f : x[(size_t)(r0 + m) * DIMS + c];
            float t = PI_F * v / BTAIL;
            f = (k < ii) ? cosf(t) : sinf(t);
        }
        sF[m * 32 + k] = f2bf(f);
    }
    __syncthreads();

    // ---- layer 1: [64,32] @ [32,800], tanh -> sH1 ----
    const unsigned short* B1i = B1 + (size_t)i * NT2 * 512;
    for (int nt = wave; nt < NT2; nt += 8) {
        Frag b;
        const v4u* bp = (const v4u*)(B1i + ((size_t)nt << 9)) + (lane << 1);
        b.u[0] = bp[0]; b.u[1] = bp[1];
        float bias = b1[(size_t)i * HID + nt * 16 + n];
        #pragma unroll
        for (int mt = 0; mt < 4; ++mt) {
            Frag a;
            const v4u* ap = (const v4u*)(sF + ((mt * 16 + n) * 32 + 16 * kh));
            a.u[0] = ap[0]; a.u[1] = ap[1];
            v8f acc = {0.f,0.f,0.f,0.f,0.f,0.f,0.f,0.f};
            acc = __builtin_amdgcn_wmma_f32_16x16x32_bf16(false, a.v, false, b.v,
                                                          (short)0, acc, false, false);
            #pragma unroll
            for (int v = 0; v < 8; ++v)
                sH1[(mt * 16 + v + 8 * kh) * HID + nt * 16 + n] =
                    f2bf(tanhf(acc[v] + bias));
        }
    }
    __syncthreads();

    // ---- layer 2: [64,800] @ [800,800], tanh -> sH2 ----
    // N-tile pairs per wave: A fragments loaded once per (kb, pair).
    const unsigned short* B2i = B2 + (size_t)i * KT2 * NT2 * 512;
    for (int p = wave; p < NT2 / 2; p += 8) {
        const int nt0 = 2 * p, nt1 = 2 * p + 1;
        v8f acc[2][4];
        #pragma unroll
        for (int j = 0; j < 2; ++j)
            #pragma unroll
            for (int mt = 0; mt < 4; ++mt)
                acc[j][mt] = (v8f){0.f,0.f,0.f,0.f,0.f,0.f,0.f,0.f};
        for (int kb = 0; kb < KT2; ++kb) {
            Frag a[4];
            #pragma unroll
            for (int mt = 0; mt < 4; ++mt) {
                const v4u* ap = (const v4u*)(sH1 + ((mt * 16 + n) * HID + kb * 32 + 16 * kh));
                a[mt].u[0] = ap[0]; a[mt].u[1] = ap[1];
            }
            Frag b0, b1;
            const v4u* bp0 = (const v4u*)(B2i + (((size_t)kb * NT2 + nt0) << 9)) + (lane << 1);
            const v4u* bp1 = (const v4u*)(B2i + (((size_t)kb * NT2 + nt1) << 9)) + (lane << 1);
            b0.u[0] = bp0[0]; b0.u[1] = bp0[1];
            b1.u[0] = bp1[0]; b1.u[1] = bp1[1];
            #pragma unroll
            for (int mt = 0; mt < 4; ++mt) {
                acc[0][mt] = __builtin_amdgcn_wmma_f32_16x16x32_bf16(
                    false, a[mt].v, false, b0.v, (short)0, acc[0][mt], false, false);
                acc[1][mt] = __builtin_amdgcn_wmma_f32_16x16x32_bf16(
                    false, a[mt].v, false, b1.v, (short)0, acc[1][mt], false, false);
            }
        }
        #pragma unroll
        for (int j = 0; j < 2; ++j) {
            const int ntj = (j == 0) ? nt0 : nt1;
            float bias = b2[(size_t)i * HID + ntj * 16 + n];
            #pragma unroll
            for (int mt = 0; mt < 4; ++mt)
                #pragma unroll
                for (int v = 0; v < 8; ++v)
                    sH2[(mt * 16 + v + 8 * kh) * HID + ntj * 16 + n] =
                        f2bf(tanhf(acc[j][mt][v] + bias));
        }
    }
    __syncthreads();

    // ---- layer 3: [64,800] @ [800,96], +bias -> sOUT (f32) ----
    const unsigned short* B3i = B3 + (size_t)i * KT2 * NT3 * 512;
    for (int nt = wave; nt < NT3; nt += 8) {
        v8f acc[4];
        #pragma unroll
        for (int mt = 0; mt < 4; ++mt)
            acc[mt] = (v8f){0.f,0.f,0.f,0.f,0.f,0.f,0.f,0.f};
        for (int kb = 0; kb < KT2; ++kb) {
            Frag b;
            const v4u* bp = (const v4u*)(B3i + (((size_t)kb * NT3 + nt) << 9)) + (lane << 1);
            b.u[0] = bp[0]; b.u[1] = bp[1];
            #pragma unroll
            for (int mt = 0; mt < 4; ++mt) {
                Frag a;
                const v4u* ap = (const v4u*)(sH2 + ((mt * 16 + n) * HID + kb * 32 + 16 * kh));
                a.u[0] = ap[0]; a.u[1] = ap[1];
                acc[mt] = __builtin_amdgcn_wmma_f32_16x16x32_bf16(
                    false, a.v, false, b.v, (short)0, acc[mt], false, false);
            }
        }
        int   col  = nt * 16 + n;
        float bias = (col < NOUT) ? b3[(size_t)i * NOUT + col] : 0.f;
        #pragma unroll
        for (int mt = 0; mt < 4; ++mt)
            #pragma unroll
            for (int v = 0; v < 8; ++v)
                sOUT[(mt * 16 + v + 8 * kh) * NOUTP + col] = acc[mt][v] + bias;
    }
    __syncthreads();

    // ---- rational-quadratic spline, one thread per row ----
    if (tid < MROWS) {
        const int    m  = tid;
        const float* o  = sOUT + m * NOUTP;
        float* cw = sSp + m * 100;   // 33 knot x
        float* ch = cw + 33;         // 33 knot y
        float* dv = ch + 33;         // 33 derivatives (also temp buffer)
        const float mdc = 1e-3f, mb = 1e-3f, scale = 1.f - mb * (float)KBINS;

        // widths: uw = 6*softmax(raw); widths = mb + scale*softmax(uw); cumsum
        float mx = -1e30f;
        for (int k = 0; k < KBINS; ++k) mx = fmaxf(mx, o[k]);
        float s = 0.f;
        for (int k = 0; k < KBINS; ++k) { float t = expf(o[k] - mx); dv[k] = t; s += t; }
        for (int k = 0; k < KBINS; ++k) dv[k] = 6.f * dv[k] / s;
        float mx2 = -1e30f;
        for (int k = 0; k < KBINS; ++k) mx2 = fmaxf(mx2, dv[k]);
        float s2 = 0.f;
        for (int k = 0; k < KBINS; ++k) { float t = expf(dv[k] - mx2); dv[k] = t; s2 += t; }
        cw[0] = -BTAIL;
        float cum = 0.f;
        for (int k = 0; k < KBINS; ++k) {
            cum += mb + scale * (dv[k] / s2);
            cw[k + 1] = 6.f * cum - BTAIL;
        }
        cw[KBINS] = BTAIL;

        // heights (same double softmax on o[32..63])
        mx = -1e30f;
        for (int k = 0; k < KBINS; ++k) mx = fmaxf(mx, o[KBINS + k]);
        s = 0.f;
        for (int k = 0; k < KBINS; ++k) { float t = expf(o[KBINS + k] - mx); dv[k] = t; s += t; }
        for (int k = 0; k < KBINS; ++k) dv[k] = 6.f * dv[k] / s;
        mx2 = -1e30f;
        for (int k = 0; k < KBINS; ++k) mx2 = fmaxf(mx2, dv[k]);
        s2 = 0.f;
        for (int k = 0; k < KBINS; ++k) { float t = expf(dv[k] - mx2); dv[k] = t; s2 += t; }
        ch[0] = -BTAIL;
        cum = 0.f;
        for (int k = 0; k < KBINS; ++k) {
            cum += mb + scale * (dv[k] / s2);
            ch[k + 1] = 6.f * cum - BTAIL;
        }
        ch[KBINS] = BTAIL;

        // derivatives: boundary = md + softplus(log(exp(1-md)-1)) == 1
        dv[0] = 1.f; dv[KBINS] = 1.f;
        for (int k = 0; k < KBINS - 1; ++k)
            dv[k + 1] = mdc + softplusf(softplusf(o[2 * KBINS + k]));

        float xv = x[(size_t)(r0 + m) * DIMS + i];
        bool  inside = (xv >= -BTAIL) && (xv <= BTAIL);
        float xc = fminf(fmaxf(xv, -BTAIL), BTAIL);
        int idx = -1;
        for (int k = 0; k <= KBINS; ++k) {
            float loc = cw[k] + ((k == KBINS) ? 1e-6f : 0.f);
            idx += (xc >= loc) ? 1 : 0;
        }
        idx = min(max(idx, 0), KBINS - 1);

        float icw = cw[idx], w  = cw[idx + 1] - icw;
        float ich = ch[idx], h  = ch[idx + 1] - ich;
        float delta = h / w;
        float d0 = dv[idx], d1 = dv[idx + 1];
        float th  = (xc - icw) / w;
        float t1m = th * (1.f - th);
        float num = h * (delta * th * th + d0 * t1m);
        float den = delta + (d0 + d1 - 2.f * delta) * t1m;
        float y   = ich + num / den;
        float dn  = delta * delta * (d1 * th * th + 2.f * delta * t1m
                                     + d0 * (1.f - th) * (1.f - th));
        float ld  = logf(dn) - 2.f * logf(den);

        z_out[(size_t)(r0 + m) * DIMS + i]    = inside ? y  : xv;
        ld_part[(size_t)i * BATCH + (r0 + m)] = inside ? ld : 0.f;
    }
}

// ---------------- deterministic log-det reduction over dims ----------------
__global__ void ld_reduce(const float* __restrict__ ld_part, float* __restrict__ out_ld) {
    int nidx = blockIdx.x * 256 + threadIdx.x;
    if (nidx < BATCH) {
        float s = 0.f;
        #pragma unroll
        for (int i = 0; i < DIMS; ++i) s += ld_part[(size_t)i * BATCH + nidx];
        out_ld[nidx] = s;
    }
}

extern "C" void kernel_launch(void* const* d_in, const int* in_sizes, int n_in,
                              void* d_out, int out_size, void* d_ws, size_t ws_size,
                              hipStream_t stream) {
    (void)in_sizes; (void)n_in; (void)out_size;
    if (ws_size < WS_BYTES) return;   // scratch too small: nothing safe to do

    const float* x  = (const float*)d_in[0];
    const float* W1 = (const float*)d_in[1];
    const float* b1 = (const float*)d_in[2];
    const float* W2 = (const float*)d_in[3];
    const float* b2 = (const float*)d_in[4];
    const float* W3 = (const float*)d_in[5];
    const float* b3 = (const float*)d_in[6];
    float* out = (float*)d_out;

    unsigned short* wsb = (unsigned short*)d_ws;
    unsigned short* B1p = wsb + OFF_B1;
    unsigned short* B2p = wsb + OFF_B2;
    unsigned short* B3p = wsb + OFF_B3;
    float* ldp = (float*)((char*)d_ws + OFF_LD_BYTES);

    pack_w1<<<(unsigned)((SZ_B1 + 255) / 256), 256, 0, stream>>>(W1, B1p);
    pack_w2<<<(unsigned)((SZ_B2 + 255) / 256), 256, 0, stream>>>(W2, B2p);
    pack_w3<<<(unsigned)((SZ_B3 + 255) / 256), 256, 0, stream>>>(W3, B3p);

    // allow >64KB dynamic LDS (CDNA5 WGP has 320KB)
    (void)hipFuncSetAttribute((const void*)nsf_main,
                              hipFuncAttributeMaxDynamicSharedMemorySize,
                              (int)LDS_BYTES);

    dim3 grid(BATCH / MROWS, DIMS);
    nsf_main<<<grid, 256, LDS_BYTES, stream>>>(x, b1, b2, b3, B1p, B2p, B3p, out, ldp);

    ld_reduce<<<BATCH / 256, 256, 0, stream>>>(ldp, out + (size_t)BATCH * DIMS);
}
